// Dca_input_stacom_45964740001824
// MI455X (gfx1250) — matrix operand order repaired and checked by reading, not compiled, not run on hardware
//
#include <hip/hip_runtime.h>
#include <hip/hip_bf16.h>

// ---------------------------------------------------------------------------
// Deformable attention for MI455X (gfx1250, wave32, WMMA).
// All heavy GEMMs run on v_wmma_f32_16x16x32_f16 with fp32 accumulation.
// Value tensor kept in fp16 (128MB) so the bilinear gather working set fits
// in the 192MB global L2.
// ---------------------------------------------------------------------------

typedef _Float16 v2h  __attribute__((ext_vector_type(2)));
typedef _Float16 v8h  __attribute__((ext_vector_type(8)));
typedef _Float16 v16h __attribute__((ext_vector_type(16)));
typedef float    v8f  __attribute__((ext_vector_type(8)));

#define WMMA_F32_F16(a, b, c) \
  __builtin_amdgcn_wmma_f32_16x16x32_f16(false, (a), false, (b), (short)0, (c), false, false)

// Constants from the reference problem
#define NPTS   40000
#define CDIM   512
#define BATCH  2
#define D0DIM  256
#define D1DIM  256
#define NHEAD  8
#define NPNT   4
#define DH     64
#define SPAT   (D0DIM * D1DIM)   // 65536 spatial positions per batch

// ---------------------------------------------------------------------------
// A-fragment for v_wmma_f32_16x16x32_f16 from a row-major [m][k] fp16 buffer.
// Layout used here (contiguous k, same as B): lane L holds row m = L%16; half = L/16;
//   VGPR0-3 hold K = half*16 + 0..7      (one contiguous v8h)
//   VGPR4-7 hold K = half*16 + 8..15     (next v8h at +8 elements)
// ---------------------------------------------------------------------------
__device__ __forceinline__ v16h make_a_frag(const _Float16* __restrict__ row, int half) {
  const v8h lo = *(const v8h*)(row + half * 16);
  const v8h hi = *(const v8h*)(row + half * 16 + 8);
  v16h r;
#pragma unroll
  for (int i = 0; i < 8; ++i) { r[i] = lo[i]; r[8 + i] = hi[i]; }
  return r;
}

// B-fragment as used here (contiguous k, same as A): lane L holds column n = L%16; lanes 0-15 hold
// K=0..15, lanes 16-31 K=16..31. With weights pre-transposed to [n][k] this
// is one contiguous 32B v16h load.

// ===========================================================================
// Small prep kernels: transpose & convert weights fp32 [k][n] -> fp16 [n][k],
// and elementwise fp32 -> fp16 conversion of q.
// ===========================================================================
__global__ void k_transpose_h(const float* __restrict__ src, _Float16* __restrict__ dst,
                              int K, int Ncols) {
  int idx = blockIdx.x * 256 + threadIdx.x;
  if (idx >= K * Ncols) return;
  int nrow = idx / K;
  int k    = idx - nrow * K;
  dst[idx] = (_Float16)src[(size_t)k * Ncols + nrow];
}

__global__ void k_convert_h(const float* __restrict__ src, _Float16* __restrict__ dst,
                            int count) {
  int idx = blockIdx.x * 256 + threadIdx.x;
  if (idx < count) dst[idx] = (_Float16)src[idx];
}

// ===========================================================================
// Value GEMM:  v[b, sp, c] = sum_k prev[b, k, sp] * Wv[k, c] + bv[c]  (fp16 out)
// prev_dense is channel-major, so A tiles (m=spatial, k=channel) are staged
// through LDS with a transpose + fp16 convert.
// Tile: 128(m) x 128(n), K chunk 32, double-buffered LDS software pipeline:
//   issue next chunk's global loads -> 8 WMMAs from current buffer ->
//   pack (v2h) + ds_store_b32 into other buffer -> single barrier.
// 8 waves: 4 in m (2 tiles each) x 2 in n (4 tiles each) => 8 acc tiles/wave.
// ===========================================================================
__global__ void __launch_bounds__(256) k_v_gemm(const float* __restrict__ prev,
                                                const _Float16* __restrict__ WvT,
                                                const float* __restrict__ bv,
                                                _Float16* __restrict__ vout) {
  __shared__ __align__(16) _Float16 As[2][128 * 40];  // [m][k], stride 40

  const int t    = threadIdx.x;
  const int lane = t & 31;
  const int wave = t >> 5;
  const int half = lane >> 4;
  const int ln   = lane & 15;

  const int bx   = blockIdx.x;
  const int b    = bx >> 11;        // 512 m-blocks * 4 n-blocks = 2048 per batch
  const int rem  = bx & 2047;
  const int mblk = rem >> 2;
  const int nblk = rem & 3;
  const int m0   = mblk * 128;      // spatial base
  const int n0   = nblk * 128;      // output-channel base
  const int wm   = wave & 3;        // wave position in m (0..3)
  const int wn   = wave >> 2;       // wave position in n (0..1)

  const size_t batch_off = (size_t)b * CDIM * SPAT;
  // Per-thread staging slot: fixed m column, 8 K-pairs (k = 4i + 2*(t>>7))
  const int sm  = t & 127;
  const int skb = (t >> 7) * 2;
  const float* gbase = prev + batch_off + m0 + sm;

  float r0[8], r1[8];
  // ---- prologue: load + store K-chunk 0 ----
#pragma unroll
  for (int i = 0; i < 8; ++i) {
    const int k = i * 4 + skb;
    r0[i] = gbase[(size_t)k * SPAT];
    r1[i] = gbase[(size_t)(k + 1) * SPAT];
  }
#pragma unroll
  for (int i = 0; i < 8; ++i) {
    const int k = i * 4 + skb;
    v2h pk; pk[0] = (_Float16)r0[i]; pk[1] = (_Float16)r1[i];
    *(v2h*)&As[0][sm * 40 + k] = pk;
  }
  __syncthreads();

  v8f acc[2][4] = {};

  for (int s = 0; s < 16; ++s) {
    const int  ks   = s * 32;
    const bool more = (s + 1 < 16);

    // Issue next chunk's global loads first (overlap HBM with WMMAs),
    // plus prefetch two chunks ahead (global_prefetch_b8).
    if (more) {
#pragma unroll
      for (int i = 0; i < 8; ++i) {
        const int k = ks + 32 + i * 4 + skb;
        r0[i] = gbase[(size_t)k * SPAT];
        r1[i] = gbase[(size_t)(k + 1) * SPAT];
        if (ks + 64 < CDIM)
          __builtin_prefetch((const void*)&gbase[(size_t)(k + 32) * SPAT], 0, 0);
      }
    }

    // Compute on current buffer
    v16h bfr[4];
#pragma unroll
    for (int j = 0; j < 4; ++j) {
      const int n = n0 + wn * 64 + j * 16 + ln;
      bfr[j] = *(const v16h*)(WvT + (size_t)n * CDIM + ks + half * 16);
    }
#pragma unroll
    for (int i = 0; i < 2; ++i) {
      v16h af = make_a_frag(&As[s & 1][(wm * 32 + i * 16 + ln) * 40], half);
#pragma unroll
      for (int j = 0; j < 4; ++j) acc[i][j] = WMMA_F32_F16(af, bfr[j], acc[i][j]);
    }

    // Convert + pack + store next chunk into the other buffer
    if (more) {
#pragma unroll
      for (int i = 0; i < 8; ++i) {
        const int k = i * 4 + skb;
        v2h pk; pk[0] = (_Float16)r0[i]; pk[1] = (_Float16)r1[i];
        *(v2h*)&As[(s + 1) & 1][sm * 40 + k] = pk;
      }
    }
    __syncthreads();
  }

  // Epilogue: add bias, convert to fp16, store (b, sp, c) layout
  float bvv[4];
#pragma unroll
  for (int j = 0; j < 4; ++j) bvv[j] = bv[n0 + wn * 64 + j * 16 + ln];
#pragma unroll
  for (int i = 0; i < 2; ++i) {
#pragma unroll
    for (int j = 0; j < 4; ++j) {
      const int n = n0 + wn * 64 + j * 16 + ln;
#pragma unroll
      for (int r = 0; r < 8; ++r) {
        const int m = m0 + wm * 32 + i * 16 + r + 8 * half;
        vout[((size_t)b * SPAT + m) * CDIM + n] = (_Float16)(acc[i][j][r] + bvv[j]);
      }
    }
  }
}

// ===========================================================================
// Projection GEMM: proj[n, 0:64]  = q @ Wo + bo   (offsets)
//                  proj[n, 64:96] = q @ Wa + ba   (attention logits)
// M=40000 (2500 tiles), N=96 (6 waves x 1 tile), K=512. No LDS needed:
// A fragments load straight from fp16 q rows, B from pre-transposed weights.
// ===========================================================================
__global__ void __launch_bounds__(192) k_proj(const _Float16* __restrict__ qh,
                                              const _Float16* __restrict__ WoaT,
                                              const float* __restrict__ bo,
                                              const float* __restrict__ ba,
                                              float* __restrict__ proj) {
  const int t    = threadIdx.x;
  const int lane = t & 31;
  const int wave = t >> 5;         // 0..5 -> n-tile
  const int half = lane >> 4;
  const int ln   = lane & 15;
  const int m0   = blockIdx.x * 16;
  const int n    = wave * 16 + ln;

  v8f acc = {};
  for (int ks = 0; ks < CDIM; ks += 32) {
    v16h af = make_a_frag(qh + (size_t)(m0 + ln) * CDIM + ks, half);
    v16h bf = *(const v16h*)(WoaT + (size_t)n * CDIM + ks + half * 16);
    acc = WMMA_F32_F16(af, bf, acc);
  }
  const float bias = (n < 64) ? bo[n] : ba[n - 64];
#pragma unroll
  for (int r = 0; r < 8; ++r) {
    const int m = m0 + r + 8 * half;
    proj[(size_t)m * 96 + n] = acc[r] + bias;
  }
}

// ===========================================================================
// Fused softmax + bilinear prep + gather. One block per point, wave per head.
// Phase 1: 32 lanes (one per head x sample point) do quad-softmax via
//          __shfl_xor, compute the 4 corner indices + attn-scaled bilinear
//          weights (validity folded in), park them in LDS.
// Phase 2: wave h accumulates 16 corners x 64 channels (2 fp16/lane) from the
//          fp16 value tensor (L2-resident), writes samp row in fp16.
// ===========================================================================
__global__ void __launch_bounds__(256) k_gather(const float* __restrict__ proj,
                                                const int* __restrict__ sp_batch,
                                                const int* __restrict__ sp_coords,
                                                const _Float16* __restrict__ vbuf,
                                                _Float16* __restrict__ samp) {
  __shared__ int   l_idx[NHEAD * NPNT * 4];
  __shared__ float l_w[NHEAD * NPNT * 4];

  const int n = blockIdx.x;
  const int t = threadIdx.x;

  if (t < 32) {
    const int h = t >> 2;
    const int p = t & 3;
    const float ox = proj[(size_t)n * 96 + h * 8 + p * 2 + 0];
    const float oy = proj[(size_t)n * 96 + h * 8 + p * 2 + 1];
    float lg = proj[(size_t)n * 96 + 64 + h * 4 + p];
    // softmax over the 4 sample points of this head (lanes form aligned quads)
    float mx = fmaxf(lg, __shfl_xor(lg, 1));
    mx       = fmaxf(mx, __shfl_xor(mx, 2));
    float e  = expf(lg - mx);
    float s  = e + __shfl_xor(e, 1);
    s        = s + __shfl_xor(s, 2);
    const float a = e / s;

    const int   bb = sp_batch[n];
    const float cx = (float)sp_coords[n * 2 + 0];
    const float cy = (float)sp_coords[n * 2 + 1];
    // loc*D - 0.5 simplifies to coord + offset - 0.5 since size == D
    const float p0 = cx + ox - 0.5f;
    const float p1 = cy + oy - 0.5f;
    const float x0f = floorf(p0), y0f = floorf(p1);
    const float fx = p0 - x0f, fy = p1 - y0f;
    const int   x0 = (int)x0f, y0 = (int)y0f;
    const float wbl[4] = {(1.f - fx) * (1.f - fy), fx * (1.f - fy),
                          (1.f - fx) * fy,         fx * fy};
    const int dx[4] = {0, 1, 0, 1};
    const int dy[4] = {0, 0, 1, 1};
#pragma unroll
    for (int c = 0; c < 4; ++c) {
      const int  xi = x0 + dx[c];
      const int  yi = y0 + dy[c];
      const bool valid = (xi >= 0) && (xi < D0DIM) && (yi >= 0) && (yi < D1DIM);
      const int  xc = min(max(xi, 0), D0DIM - 1);
      const int  yc = min(max(yi, 0), D1DIM - 1);
      l_idx[t * 4 + c] = (bb << 16) | (xc << 8) | yc;   // flat (b, x, y)
      l_w[t * 4 + c]   = valid ? (a * wbl[c]) : 0.f;
    }
  }
  __syncthreads();

  const int h    = t >> 5;           // wave id == head
  const int lane = t & 31;
  const int li   = l_idx[h * 16 + (lane & 15)];
  const float lw = l_w[h * 16 + (lane & 15)];

  float ax = 0.f, ay = 0.f;
  const size_t cbase = (size_t)h * DH + 2 * lane;   // 2 channels per lane
#pragma unroll
  for (int j = 0; j < 16; ++j) {
    const int   idx = __shfl(li, j);
    const float w   = __shfl(lw, j);
    const v2h d = *(const v2h*)(vbuf + (size_t)idx * CDIM + cbase);
    ax += w * (float)d[0];
    ay += w * (float)d[1];
  }
  v2h o;
  o[0] = (_Float16)ax;
  o[1] = (_Float16)ay;
  *(v2h*)(samp + (size_t)n * CDIM + cbase) = o;
}

// ===========================================================================
// Output GEMM + residual: out = samp @ Wout + bout + sp_features.
// M=40000 (2500 tiles), N=512 (8 waves x 4 tiles), K=512. Direct global
// fragment loads (samp rows fp16, WoutT [n][k] fp16 cached in L2).
// ===========================================================================
__global__ void __launch_bounds__(256) k_out_gemm(const _Float16* __restrict__ samp,
                                                  const _Float16* __restrict__ WoutT,
                                                  const float* __restrict__ bout,
                                                  const float* __restrict__ spf,
                                                  float* __restrict__ out) {
  const int t    = threadIdx.x;
  const int lane = t & 31;
  const int wave = t >> 5;
  const int half = lane >> 4;
  const int ln   = lane & 15;
  const int m0   = blockIdx.x * 16;
  const int nb   = wave * 64;

  v8f acc[4] = {};
  for (int ks = 0; ks < CDIM; ks += 32) {
    v16h af = make_a_frag(samp + (size_t)(m0 + ln) * CDIM + ks, half);
#pragma unroll
    for (int j = 0; j < 4; ++j) {
      v16h bf = *(const v16h*)(WoutT + (size_t)(nb + j * 16 + ln) * CDIM + ks + half * 16);
      acc[j] = WMMA_F32_F16(af, bf, acc[j]);
    }
  }
#pragma unroll
  for (int j = 0; j < 4; ++j) {
    const int   nn = nb + j * 16 + ln;
    const float bb = bout[nn];
#pragma unroll
    for (int r = 0; r < 8; ++r) {
      const int    m = m0 + r + 8 * half;
      const size_t o = (size_t)m * CDIM + nn;
      out[o] = acc[j][r] + bb + spf[o];
    }
  }
}

// ===========================================================================
// Host launcher
// ===========================================================================
extern "C" void kernel_launch(void* const* d_in, const int* in_sizes, int n_in,
                              void* d_out, int out_size, void* d_ws, size_t ws_size,
                              hipStream_t stream) {
  (void)in_sizes; (void)n_in; (void)out_size; (void)ws_size;

  const float* spf       = (const float*)d_in[0];
  const int*   sp_batch  = (const int*)d_in[1];
  const int*   sp_coords = (const int*)d_in[2];
  const float* prev      = (const float*)d_in[3];
  const float* Wv        = (const float*)d_in[4];
  const float* bv        = (const float*)d_in[5];
  const float* Wo        = (const float*)d_in[6];
  const float* bo        = (const float*)d_in[7];
  const float* Wa        = (const float*)d_in[8];
  const float* ba        = (const float*)d_in[9];
  const float* Wout      = (const float*)d_in[10];
  const float* bout      = (const float*)d_in[11];
  float*       out       = (float*)d_out;

  // Workspace layout (all offsets 256B aligned)
  char* ws = (char*)d_ws;
  _Float16* vbuf  = (_Float16*)(ws + 0);           // 2*65536*512 fp16 = 128MB
  _Float16* WvT   = (_Float16*)(ws + 134217728);   // 512KB
  _Float16* WoutT = (_Float16*)(ws + 134742016);   // 512KB
  _Float16* WoaT  = (_Float16*)(ws + 135266304);   // 96x512 fp16 (128KB slot)
  _Float16* qh    = (_Float16*)(ws + 135397376);   // 40000x512 fp16 ~= 40MB
  float*    proj  = (float*)   (ws + 176357376);   // 40000x96  f32  ~= 15MB
  _Float16* samp  = (_Float16*)(ws + 191717376);   // 40000x512 fp16 ~= 40MB

  // Weight transposes to [n][k] fp16 + q conversion
  k_transpose_h<<<(512 * 512 + 255) / 256, 256, 0, stream>>>(Wv, WvT, 512, 512);
  k_transpose_h<<<(512 * 512 + 255) / 256, 256, 0, stream>>>(Wout, WoutT, 512, 512);
  k_transpose_h<<<(64 * 512 + 255) / 256, 256, 0, stream>>>(Wo, WoaT, 512, 64);
  k_transpose_h<<<(32 * 512 + 255) / 256, 256, 0, stream>>>(Wa, WoaT + 64 * 512, 512, 32);
  k_convert_h<<<(NPTS * CDIM + 255) / 256, 256, 0, stream>>>(spf, qh, NPTS * CDIM);

  // Value projection GEMM (dominant): 2 batches * 512 m-blocks * 4 n-blocks
  k_v_gemm<<<BATCH * 512 * 4, 256, 0, stream>>>(prev, WvT, bv, vbuf);

  // Offset / attention-logit projection
  k_proj<<<NPTS / 16, 192, 0, stream>>>(qh, WoaT, bo, ba, proj);

  // Softmax + bilinear gather (fused)
  k_gather<<<NPTS, 256, 0, stream>>>(proj, sp_batch, sp_coords, vbuf, samp);

  // Output projection + bias + residual
  k_out_gemm<<<NPTS / 16, 256, 0, stream>>>(samp, WoutT, bout, spf, out);
}
